// RegionalAttentionModule_37941741093296
// MI455X (gfx1250) — matrix operand order repaired and checked by reading, not compiled, not run on hardware
//
#include <hip/hip_runtime.h>

typedef __attribute__((ext_vector_type(16))) __bf16 bf16x16;
typedef __attribute__((ext_vector_type(2)))  __bf16 bf16x2;
typedef __attribute__((ext_vector_type(8)))  float  f32x8;
typedef __attribute__((ext_vector_type(4)))  float  f32x4;
typedef __attribute__((ext_vector_type(4)))  unsigned int u32x4;

union Frag {
  bf16x16 v;
  unsigned short u[16];
  u32x4 q[2];
};

// f32 -> bf16 via native cast: clang lowers this inline for amdgcn and picks
// the HW cvt (v_cvt_pk_bf16_f32 for adjacent pairs) when the target has it.
__device__ __forceinline__ unsigned short f32_to_bf16(float f) {
  union { __bf16 h; unsigned short u; } c;
  c.h = (__bf16)f;
  return c.u;
}

#if __has_builtin(__builtin_amdgcn_cvt_pk_bf16_f32)
__device__ __forceinline__ unsigned int pk_bf16(float a, float b) {
  union { bf16x2 v; unsigned int u; } c;
  c.v = __builtin_amdgcn_cvt_pk_bf16_f32(a, b);
  return c.u;
}
#else
__device__ __forceinline__ unsigned int pk_bf16(float a, float b) {
  union { bf16x2 v; unsigned int u; } c;
  bf16x2 t;
  t.x = (__bf16)a;
  t.y = (__bf16)b;
  c.v = t;
  return c.u;
}
#endif

static constexpr int Bb = 2, NN = 2048, CC = 1024, HH = 16, DD = 64;
static constexpr int MM = Bb * NN;  // 4096 rows of x

// ---------------------------------------------------------------------------
// Kernel 1: Y = x @ W^T + b  ->  bf16 [B,H,N,d]   (grid.z picks Q / K / V)
// Block tile: 64 (M) x 128 (O), 256 threads = 8 waves, each wave 32x32.
// K-step 64 = two v_wmma_f32_16x16x32_bf16 K-chunks per stage (8 WMMA/stage).
// ---------------------------------------------------------------------------
__global__ __launch_bounds__(256) void qkv_gemm_bf16(
    const float* __restrict__ x,
    const float* __restrict__ Wq, const float* __restrict__ bq,
    const float* __restrict__ Wk, const float* __restrict__ bk,
    const float* __restrict__ Wv, const float* __restrict__ bv,
    unsigned short* __restrict__ Qb, unsigned short* __restrict__ Kb,
    unsigned short* __restrict__ Vb) {
  const int z = blockIdx.z;
  const float* W    = (z == 0) ? Wq : (z == 1) ? Wk : Wv;
  const float* bias = (z == 0) ? bq : (z == 1) ? bk : bv;
  unsigned short* outp = (z == 0) ? Qb : (z == 1) ? Kb : Vb;

  const int m0 = blockIdx.x * 64;
  const int o0 = blockIdx.y * 128;
  const int tid  = threadIdx.x;
  const int w    = tid >> 5;
  const int L    = tid & 31;
  const int half = L >> 4;
  const int lm   = L & 15;
  const int mw = (w >> 2) * 32;  // 0 / 32
  const int ow = (w & 3) * 32;   // 0..96

  // A tile: 64 rows x 64 k (bf16), row stride 72 (16B-aligned rows)
  // B tile transposed [k][o]: 64 x 128, row stride 136
  __shared__ __align__(16) unsigned short Alds[64 * 72];
  __shared__ __align__(16) unsigned short Blds[64 * 136];

  f32x8 acc[2][2];
#pragma unroll
  for (int i = 0; i < 2; ++i)
#pragma unroll
    for (int j = 0; j < 2; ++j) acc[i][j] = (f32x8)(0.0f);

  for (int kb = 0; kb < CC; kb += 64) {
    // ---- stage A: x[m0..m0+63, kb..kb+63] -> bf16 LDS (packed dwords) ----
    {
      const int r = tid >> 2, cq = (tid & 3) * 16;
      const float* src = x + (size_t)(m0 + r) * CC + kb + cq;
      f32x4 f0 = *(const f32x4*)src;
      f32x4 f1 = *(const f32x4*)(src + 4);
      f32x4 f2 = *(const f32x4*)(src + 8);
      f32x4 f3 = *(const f32x4*)(src + 12);
      u32x4 w0 = {pk_bf16(f0.x, f0.y), pk_bf16(f0.z, f0.w),
                  pk_bf16(f1.x, f1.y), pk_bf16(f1.z, f1.w)};
      u32x4 w1 = {pk_bf16(f2.x, f2.y), pk_bf16(f2.z, f2.w),
                  pk_bf16(f3.x, f3.y), pk_bf16(f3.z, f3.w)};
      *(u32x4*)&Alds[r * 72 + cq] = w0;
      *(u32x4*)&Alds[r * 72 + cq + 8] = w1;
      if (kb + 64 < CC) __builtin_prefetch(src + 64, 0, 1);
    }
    // ---- stage B: W[o0+2t_o .. +1, kb+kq..+15] -> LDS [k][o], dword pairs ----
    {
      const int o  = (tid >> 2) * 2;      // even column pair 0..126
      const int kq = (tid & 3) * 16;
      const float* s0 = W + (size_t)(o0 + o) * CC + kb + kq;
      const float* s1 = s0 + CC;
      f32x4 a[4], b[4];
#pragma unroll
      for (int c = 0; c < 4; ++c) {
        a[c] = *(const f32x4*)(s0 + 4 * c);
        b[c] = *(const f32x4*)(s1 + 4 * c);
      }
#pragma unroll
      for (int jj = 0; jj < 16; ++jj) {
        const unsigned int u = pk_bf16(a[jj >> 2][jj & 3], b[jj >> 2][jj & 3]);
        *(unsigned int*)&Blds[(kq + jj) * 136 + o] = u;
      }
      if (kb + 64 < CC) __builtin_prefetch(s0 + 64, 0, 1);
    }
    __syncthreads();

    // ---- fragments + WMMA (two K=32 chunks) ----
    Frag a[2][2], bf[2][2];
#pragma unroll
    for (int i = 0; i < 2; ++i) {
      const int row = mw + 16 * i + lm;
#pragma unroll
      for (int kc = 0; kc < 2; ++kc) {
        a[i][kc].q[0] = *(const u32x4*)&Alds[row * 72 + kc * 32 + 8 * half];
        a[i][kc].q[1] = *(const u32x4*)&Alds[row * 72 + kc * 32 + 16 + 8 * half];
      }
    }
#pragma unroll
    for (int j = 0; j < 2; ++j)
#pragma unroll
      for (int kc = 0; kc < 2; ++kc) {
#pragma unroll
        for (int s = 0; s < 16; ++s)
          bf[j][kc].u[s] = Blds[(kc * 32 + 8 * half + s + (s & 8)) * 136 + ow + 16 * j + lm];
      }
#pragma unroll
    for (int kc = 0; kc < 2; ++kc)
#pragma unroll
      for (int i = 0; i < 2; ++i)
#pragma unroll
        for (int j = 0; j < 2; ++j)
          acc[i][j] = __builtin_amdgcn_wmma_f32_16x16x32_bf16(
              false, a[i][kc].v, false, bf[j][kc].v, (short)0, acc[i][j],
              false, false);
    __syncthreads();
  }

  // ---- epilogue: +bias, bf16 store into [B,H,N,d] ----
#pragma unroll
  for (int i = 0; i < 2; ++i)
#pragma unroll
    for (int j = 0; j < 2; ++j)
#pragma unroll
      for (int e = 0; e < 8; ++e) {
        const int m = m0 + mw + 16 * i + 8 * half + e;
        const int o = o0 + ow + 16 * j + lm;
        const float v = acc[i][j][e] + bias[o];
        const int bIdx = m >> 11, nr = m & (NN - 1);
        const int hh = o >> 6, dd = o & 63;
        outp[((((size_t)bIdx * HH + hh) * NN + nr) << 6) + dd] = f32_to_bf16(v);
      }
}

// ---------------------------------------------------------------------------
// Kernel 2: flash attention per (b,h). Block = 128 threads (4 waves),
// each wave owns 16 queries; keys streamed in tiles of 32.
// S = Q K^T (bf16 WMMA, fp32 acc), online softmax, O += P V (bf16 WMMA).
// ---------------------------------------------------------------------------
__global__ __launch_bounds__(128) void flash_attn_bf16(
    const unsigned short* __restrict__ Qb, const unsigned short* __restrict__ Kb,
    const unsigned short* __restrict__ Vb, const float* __restrict__ rmask,
    const float* __restrict__ rweight, float* __restrict__ out) {
  const int tid  = threadIdx.x;
  const int w    = tid >> 5;
  const int L    = tid & 31;
  const int half = L >> 4;
  const int lm   = L & 15;
  const int bh = blockIdx.x;
  const int b  = bh >> 4;
  const int h  = bh & 15;
  const int qw = blockIdx.y * 64 + w * 16;
  const float scale = 0.125f;  // d^-0.5, d=64
  const float rw = rweight[0];

  __shared__ __align__(16) unsigned short Kt[64 * 40];       // [d][key], stride 40
  __shared__ __align__(16) unsigned short Pt[4 * 16 * 40];   // per-wave P tile
  unsigned short* Ptw = &Pt[w * 640];

  // per-row additive bias (cancels in softmax; kept for faithfulness)
  float bias_r[8];
#pragma unroll
  for (int i = 0; i < 8; ++i)
    bias_r[i] = rmask[b * NN + qw + i + 8 * half] * rw;

  // Q A-fragments for this wave's 16 queries (d = 0..31, 32..63)
  Frag aq[2];
  {
    const size_t qbase = ((size_t)bh * NN + qw + lm) * DD;
    aq[0].q[0] = *(const u32x4*)(Qb + qbase + 8 * half);
    aq[0].q[1] = *(const u32x4*)(Qb + qbase + 16 + 8 * half);
    aq[1].q[0] = *(const u32x4*)(Qb + qbase + 32 + 8 * half);
    aq[1].q[1] = *(const u32x4*)(Qb + qbase + 48 + 8 * half);
  }

  float Mrow[8], Srow[8];
  f32x8 oacc[4];
#pragma unroll
  for (int i = 0; i < 8; ++i) { Mrow[i] = -3.0e38f; Srow[i] = 0.0f; }
#pragma unroll
  for (int d = 0; d < 4; ++d) oacc[d] = (f32x8)(0.0f);

  for (int kt = 0; kt < NN; kt += 32) {
    // ---- stage K^T tile: [key][d] -> [d][key] LDS, packed key-pair dwords ----
    {
      const int kp = tid >> 3;          // key pair 0..15
      const int dq = (tid & 7) * 8;     // d chunk 0..56
      const size_t src = ((size_t)bh * NN + kt + 2 * kp) * DD + dq;
      union { u32x4 q; unsigned short u[8]; } c0, c1;
      c0.q = *(const u32x4*)(Kb + src);        // key 2kp
      c1.q = *(const u32x4*)(Kb + src + DD);   // key 2kp+1
#pragma unroll
      for (int j = 0; j < 8; ++j) {
        const unsigned int u =
            (unsigned int)c0.u[j] | ((unsigned int)c1.u[j] << 16);
        *(unsigned int*)&Kt[(dq + j) * 40 + 2 * kp] = u;
      }
      if (kt + 32 < NN) __builtin_prefetch(Kb + src + 32 * DD, 0, 1);
    }
    __syncthreads();

    // ---- S = Q K^T : 16 q x 32 keys per wave ----
    f32x8 s[2];
    s[0] = (f32x8)(0.0f); s[1] = (f32x8)(0.0f);
#pragma unroll
    for (int dc = 0; dc < 2; ++dc) {
      const int drow = dc * 32 + L;  // unused; B-frag lane = key col, d per element
#pragma unroll
      for (int j = 0; j < 2; ++j) {
        Frag bk;
#pragma unroll
        for (int s = 0; s < 16; ++s)
          bk.u[s] = Kt[(dc * 32 + 8 * half + s + (s & 8)) * 40 + 16 * j + lm];
        s[j] = __builtin_amdgcn_wmma_f32_16x16x32_bf16(
            false, aq[dc].v, false, bk.v, (short)0, s[j], false, false);
      }
    }

    // ---- online softmax (rows replicated across 16-lane halves) ----
#pragma unroll
    for (int i = 0; i < 8; ++i) {
      float s0 = s[0][i] * scale + bias_r[i];
      float s1 = s[1][i] * scale + bias_r[i];
      float mx = fmaxf(s0, s1);
#pragma unroll
      for (int msk = 1; msk < 16; msk <<= 1)
        mx = fmaxf(mx, __shfl_xor(mx, msk, 32));
      const float mnew = fmaxf(Mrow[i], mx);
      const float corr = __expf(Mrow[i] - mnew);
      const float p0 = __expf(s0 - mnew);
      const float p1 = __expf(s1 - mnew);
      float rs = p0 + p1;
#pragma unroll
      for (int msk = 1; msk < 16; msk <<= 1)
        rs += __shfl_xor(rs, msk, 32);
      Srow[i] = Srow[i] * corr + rs;
      Mrow[i] = mnew;
#pragma unroll
      for (int d = 0; d < 4; ++d) oacc[d][i] *= corr;
      // C-layout -> LDS (row = q, col = key) for A-frag re-layout
      Ptw[(i + 8 * half) * 40 + lm]      = f32_to_bf16(p0);
      Ptw[(i + 8 * half) * 40 + 16 + lm] = f32_to_bf16(p1);
    }

    // ---- P as A-fragment ----
    Frag ap;
    ap.q[0] = *(const u32x4*)&Ptw[lm * 40 + 8 * half];
    ap.q[1] = *(const u32x4*)&Ptw[lm * 40 + 16 + 8 * half];

    // ---- O += P V : V B-frags straight from global (lane = d col, elements = key rows) ----
#pragma unroll
    for (int d = 0; d < 4; ++d) {
      Frag bv;
      const size_t vsrc = ((size_t)bh * NN + kt) * DD + d * 16 + lm;
#pragma unroll
      for (int s = 0; s < 16; ++s)
        bv.u[s] = Vb[vsrc + (size_t)(8 * half + s + (s & 8)) * DD];
      oacc[d] = __builtin_amdgcn_wmma_f32_16x16x32_bf16(
          false, ap.v, false, bv.v, (short)0, oacc[d], false, false);
    }
    __syncthreads();
  }

  // ---- normalize + fp32 store to [B,N,C] ----
#pragma unroll
  for (int i = 0; i < 8; ++i) {
    const float inv = 1.0f / Srow[i];
    const size_t row = (size_t)(b * NN + qw + i + 8 * half) * CC + h * DD;
#pragma unroll
    for (int d = 0; d < 4; ++d)
      out[row + d * 16 + lm] = oacc[d][i] * inv;
  }
}

// ---------------------------------------------------------------------------
extern "C" void kernel_launch(void* const* d_in, const int* in_sizes, int n_in,
                              void* d_out, int out_size, void* d_ws, size_t ws_size,
                              hipStream_t stream) {
  const float* x  = (const float*)d_in[0];
  const float* rm = (const float*)d_in[1];
  const float* Wq = (const float*)d_in[2];
  const float* bq = (const float*)d_in[3];
  const float* Wk = (const float*)d_in[4];
  const float* bk = (const float*)d_in[5];
  const float* Wv = (const float*)d_in[6];
  const float* bv = (const float*)d_in[7];
  const float* rw = (const float*)d_in[8];
  float* out = (float*)d_out;

  const size_t tensor_elems = (size_t)MM * CC;  // 4096*1024 bf16 each
  unsigned short* Qb = (unsigned short*)d_ws;
  unsigned short* Kb = Qb + tensor_elems;
  unsigned short* Vb = Kb + tensor_elems;

  dim3 g1(MM / 64, CC / 128, 3);
  qkv_gemm_bf16<<<g1, dim3(256), 0, stream>>>(x, Wq, bq, Wk, bk, Wv, bv, Qb, Kb, Vb);

  dim3 g2(Bb * HH, NN / 64);
  flash_attn_bf16<<<g2, dim3(128), 0, stream>>>(Qb, Kb, Vb, rm, rw, out);
}
